// Attention_14362370637857
// MI455X (gfx1250) — matrix operand order repaired and checked by reading, not compiled, not run on hardware
//
#include <hip/hip_runtime.h>

// ---------------------------------------------------------------------------
// CDNA5 (gfx1250, wave32) attention pipeline.
//  - all matmuls via v_wmma_f32_16x16x32_bf16 (fp32 accumulate)
//  - all global->LDS tile movement via global_load_async_to_lds_b128
//    (ASYNCcnt path), double buffered so DMA overlaps WMMA compute
//  - every matrix operand kept in weight-style [N][K] layout so WMMA
//    fragment gathers are aligned dword LDS loads (merge into ds_load_b128)
// ---------------------------------------------------------------------------

#define DEV __device__ __forceinline__

typedef __attribute__((ext_vector_type(16))) __bf16 v16bf;
typedef __attribute__((ext_vector_type(8)))  float  v8f;

union FragA { unsigned int u[8]; v16bf v; };

DEV unsigned short f2bf(float f) {
  unsigned int u = __builtin_bit_cast(unsigned int, f);
  u = (u + 0x7FFFu + ((u >> 16) & 1u)) >> 16;   // round-to-nearest-even
  return (unsigned short)u;
}

// ISA 7.12.2: 16-bit A-matrix 16x32 per-lane K offsets (even base of a pair)
DEV int ka_off(int i, int sel) { return (i < 4 ? 2 * i : 16 + 2 * (i - 4)) + 8 * sel; }
// 16-bit B-matrix 32x16 per-lane K offsets (same K order as A)
DEV int kb_off(int i, int sel) { return (i < 4 ? 2 * i : 16 + 2 * (i - 4)) + 8 * sel; }

// Generic->LDS aperture: low 32 bits of a generic LDS address are the LDS
// byte offset (ISA 10.2).
DEV unsigned lds_off(const void* p) { return (unsigned)(unsigned long long)p; }

// Async DMA: 16 bytes per lane, global -> LDS, tracked by ASYNCcnt.
DEV void async_load_b128(unsigned lds_byte_off, const void* gaddr) {
  asm volatile("global_load_async_to_lds_b128 %0, %1, off"
               :: "v"(lds_byte_off), "v"((unsigned long long)gaddr)
               : "memory");
}
DEV void wait_async0() { asm volatile("s_wait_asynccnt 0x0" ::: "memory"); }

__global__ __launch_bounds__(256) void cast_f32_to_bf16(const float* __restrict__ src,
                                                        unsigned short* __restrict__ dst,
                                                        int n) {
  int i = blockIdx.x * blockDim.x + threadIdx.x;
  int stride = gridDim.x * blockDim.x;
  for (; i < n; i += stride) dst[i] = f2bf(src[i]);
}

// W_O [c=n*64+h][d] fp32 -> W_O^T [d][c] bf16 (so O-proj is weight-style [N][K])
__global__ __launch_bounds__(256) void cast_transpose_wo(const float* __restrict__ Wo,
                                                         unsigned short* __restrict__ Wt) {
  int i = blockIdx.x * blockDim.x + threadIdx.x;
  int stride = gridDim.x * blockDim.x;
  for (; i < 2048 * 2048; i += stride) {
    int c = i & 2047, d = i >> 11;                 // write-linear over Wt[d][c]
    Wt[i] = f2bf(Wo[(size_t)c * 2048 + d]);
  }
}

// V region of QKV [b*2048+s][2560 + kv*64 + h] -> Vt[((b*8+kv)*64+h)][s] bf16
__global__ __launch_bounds__(256) void transpose_v(const unsigned short* __restrict__ QKV,
                                                   unsigned short* __restrict__ Vt) {
  int i = blockIdx.x * blockDim.x + threadIdx.x;
  int stride = gridDim.x * blockDim.x;
  for (; i < 2 * 512 * 2048; i += stride) {
    int s = i & 2047;
    int c = (i >> 11) & 511;                       // kv*64+h
    int b = i >> 20;
    Vt[(size_t)(b * 512 + c) * 2048 + s] =
        QKV[(size_t)(b * 2048 + s) * 3072 + 2560 + c];
  }
}

// ---------------------------------------------------------------------------
// WMMA GEMM, weight-style B:  C[m,n] = sum_k A[m,k] * B[n,k]
// A [M][K] bf16, B [N][K] bf16.  256 thr = 8 waves -> 64x128 block tile,
// each wave 32x32 (2x2 frags), K step 32, double-buffered async LDS tiles.
// ---------------------------------------------------------------------------
template <bool OUT_BF16>
__global__ __launch_bounds__(256) void gemm_wmma_bf16(const unsigned short* __restrict__ A,
                                                      const unsigned short* __restrict__ B,
                                                      void* __restrict__ C,
                                                      int K, int ldc) {
  __shared__ unsigned short As[2][64 * 32];
  __shared__ unsigned short Bs[2][128 * 32];

  const int tid  = threadIdx.x;
  const int lane = tid & 31;
  const int w    = tid >> 5;
  const int wm   = (w & 1) * 32;
  const int wn   = (w >> 1) * 32;
  const int m0   = blockIdx.y * 64;
  const int n0   = blockIdx.x * 128;
  const int sel  = lane >> 4;
  const int l15  = lane & 15;

  auto issue_tile = [&](int kk, int buf) {
    unsigned aoff = lds_off(&As[buf][0]);
    unsigned boff = lds_off(&Bs[buf][0]);
    {                                                // A: 64 rows x 64B
      int r = tid >> 2, c = tid & 3;
      async_load_b128(aoff + tid * 16,
                      (const char*)A + ((size_t)(m0 + r) * K + kk) * 2 + c * 16);
    }
#pragma unroll
    for (int i = 0; i < 2; ++i) {                    // B: 128 rows x 64B
      int idx = tid + i * 256;
      int r = idx >> 2, c = idx & 3;
      async_load_b128(boff + idx * 16,
                      (const char*)B + ((size_t)(n0 + r) * K + kk) * 2 + c * 16);
    }
  };

  const v8f z8 = {0.f, 0.f, 0.f, 0.f, 0.f, 0.f, 0.f, 0.f};
  v8f acc[2][2];
  acc[0][0] = z8; acc[0][1] = z8; acc[1][0] = z8; acc[1][1] = z8;

  issue_tile(0, 0);
  int ib = 0;
  for (int kk = 0; kk < K; kk += 32, ib ^= 1) {
    wait_async0();
    __syncthreads();
    if (kk + 32 < K) issue_tile(kk + 32, ib ^ 1);

    FragA a[2], bf[2];
#pragma unroll
    for (int mi = 0; mi < 2; ++mi)
#pragma unroll
      for (int i = 0; i < 8; ++i)
        a[mi].u[i] =
            *(const unsigned int*)&As[ib][(wm + mi * 16 + l15) * 32 + ka_off(i, sel)];
#pragma unroll
    for (int ni = 0; ni < 2; ++ni)
#pragma unroll
      for (int i = 0; i < 8; ++i)
        bf[ni].u[i] =
            *(const unsigned int*)&Bs[ib][(wn + ni * 16 + l15) * 32 + kb_off(i, sel)];

#pragma unroll
    for (int mi = 0; mi < 2; ++mi)
#pragma unroll
      for (int ni = 0; ni < 2; ++ni)
        acc[mi][ni] = __builtin_amdgcn_wmma_f32_16x16x32_bf16(
            false, a[mi].v, false, bf[ni].v, (short)0, acc[mi][ni], false, false);
  }

  // C/D layout: elem j -> row j + 8*sel, col = l15
#pragma unroll
  for (int mi = 0; mi < 2; ++mi)
#pragma unroll
    for (int ni = 0; ni < 2; ++ni)
#pragma unroll
      for (int j = 0; j < 8; ++j) {
        int gm = m0 + wm + mi * 16 + j + 8 * sel;
        int gn = n0 + wn + ni * 16 + l15;
        float val = acc[mi][ni][j];
        if (OUT_BF16)
          ((unsigned short*)C)[(size_t)gm * ldc + gn] = f2bf(val);
        else
          ((float*)C)[(size_t)gm * ldc + gn] = val;
      }
}

// ---------------------------------------------------------------------------
// Flash attention, causal, GQA 4:1.  Block = (q-tile of 128, head, batch),
// 8 waves x 16 q-rows.  K and V^T streamed in 64-wide tiles via async DMA,
// double buffered.
// ---------------------------------------------------------------------------
__global__ __launch_bounds__(256) void flash_attn_wmma(const unsigned short* __restrict__ QKV,
                                                       const unsigned short* __restrict__ Vt,
                                                       unsigned short* __restrict__ Out) {
  __shared__ unsigned short Qs[128 * 64];
  __shared__ unsigned short Ks[2][64 * 64];
  __shared__ unsigned short Vs[2][64 * 64];   // holds V^T tile: [h][k]
  __shared__ unsigned short Ps[8 * 16 * 64];  // per-wave P staging (C->A relayout)

  const int tid  = threadIdx.x;
  const int lane = tid & 31;
  const int w    = tid >> 5;
  const int sel  = lane >> 4;
  const int l15  = lane & 15;

  const int qb   = blockIdx.x;   // 0..15
  const int head = blockIdx.y;   // 0..31
  const int b    = blockIdx.z;   // 0..1
  const int kvh  = head >> 2;

  const size_t rowQ = (size_t)(b * 2048 + qb * 128);

  auto issue_kv = [&](int kt, int buf) {
    int kbase = kt * 64;
    unsigned koff = lds_off(&Ks[buf][0]);
    unsigned voff = lds_off(&Vs[buf][0]);
#pragma unroll
    for (int i = 0; i < 2; ++i) {
      int idx = tid + i * 256;                 // 0..511, 64 rows x 128B each tile
      int r = idx >> 3, c = idx & 7;
      async_load_b128(koff + idx * 16,
                      (const char*)QKV +
                          ((size_t)(b * 2048 + kbase + r) * 3072 + 2048 + kvh * 64) * 2 +
                          c * 16);
      async_load_b128(voff + idx * 16,
                      (const char*)Vt +
                          ((size_t)(b * 512 + kvh * 64 + r) * 2048 + kbase) * 2 + c * 16);
    }
  };

  // Q tile: 128 rows x 128B via async DMA
#pragma unroll
  for (int i = 0; i < 4; ++i) {
    int idx = tid + i * 256;
    int r = idx >> 3, c = idx & 7;
    async_load_b128(lds_off(Qs) + idx * 16,
                    (const char*)QKV + ((rowQ + r) * 3072 + head * 64) * 2 + c * 16);
  }
  const int ktiles = (qb + 1) * 2;            // causal: skip tiles past diagonal
  issue_kv(0, 0);
  wait_async0();
  __syncthreads();

  // this wave's Q A-fragments (16 rows x 64 = two K-steps), loop-invariant
  FragA qa[2];
#pragma unroll
  for (int ks = 0; ks < 2; ++ks)
#pragma unroll
    for (int i = 0; i < 8; ++i)
      qa[ks].u[i] =
          *(const unsigned int*)&Qs[(w * 16 + l15) * 64 + ka_off(i, sel) + ks * 32];

  const v8f z8 = {0.f, 0.f, 0.f, 0.f, 0.f, 0.f, 0.f, 0.f};
  v8f o[4];
  o[0] = z8; o[1] = z8; o[2] = z8; o[3] = z8;
  float rmax[8], rsum[8];
#pragma unroll
  for (int j = 0; j < 8; ++j) { rmax[j] = -3.0e38f; rsum[j] = 0.f; }

  const int qrow0 = qb * 128 + w * 16 + 8 * sel;   // + j = this elem's q index

  int ib = 0;
  for (int kt = 0; kt < ktiles; ++kt, ib ^= 1) {
    const int kbase = kt * 64;
    if (kt + 1 < ktiles) issue_kv(kt + 1, ib ^ 1);

    // scores S = Q K^T : 16 q-rows x 64 k-cols, 8 WMMAs
    v8f s[4];
    s[0] = z8; s[1] = z8; s[2] = z8; s[3] = z8;
#pragma unroll
    for (int c4 = 0; c4 < 4; ++c4)
#pragma unroll
      for (int ks = 0; ks < 2; ++ks) {
        FragA kf;
#pragma unroll
        for (int i = 0; i < 8; ++i)
          kf.u[i] = *(const unsigned int*)
              &Ks[ib][(c4 * 16 + l15) * 64 + kb_off(i, sel) + ks * 32];
        s[c4] = __builtin_amdgcn_wmma_f32_16x16x32_bf16(false, qa[ks].v, false, kf.v,
                                                        (short)0, s[c4], false, false);
      }

    // scale 1/sqrt(64) + causal mask
#pragma unroll
    for (int c4 = 0; c4 < 4; ++c4) {
      int kg = kbase + c4 * 16 + l15;
#pragma unroll
      for (int j = 0; j < 8; ++j) {
        float v = s[c4][j] * 0.125f;
        s[c4][j] = (kg > qrow0 + j) ? -3.0e38f : v;
      }
    }

    // online softmax: strip max (4 frags + 16-lane half-wave reduction)
    float corr[8];
#pragma unroll
    for (int j = 0; j < 8; ++j) {
      float t = fmaxf(fmaxf(s[0][j], s[1][j]), fmaxf(s[2][j], s[3][j]));
#pragma unroll
      for (int off = 8; off >= 1; off >>= 1) t = fmaxf(t, __shfl_xor(t, off, 32));
      float nm = fmaxf(rmax[j], t);
      corr[j] = __expf(rmax[j] - nm);
      rmax[j] = nm;
      rsum[j] *= corr[j];
    }
#pragma unroll
    for (int ht = 0; ht < 4; ++ht)
#pragma unroll
      for (int j = 0; j < 8; ++j) o[ht][j] *= corr[j];

    // P = exp(S - max); stage to LDS in [m][k] order for A-layout reload
    float lsum[8];
#pragma unroll
    for (int j = 0; j < 8; ++j) lsum[j] = 0.f;
#pragma unroll
    for (int c4 = 0; c4 < 4; ++c4)
#pragma unroll
      for (int j = 0; j < 8; ++j) {
        float p = __expf(s[c4][j] - rmax[j]);
        lsum[j] += p;
        Ps[w * 1024 + (j + 8 * sel) * 64 + c4 * 16 + l15] = f2bf(p);
      }
#pragma unroll
    for (int j = 0; j < 8; ++j) {
      float t = lsum[j];
#pragma unroll
      for (int off = 8; off >= 1; off >>= 1) t += __shfl_xor(t, off, 32);
      rsum[j] += t;
    }
    // wave-local LDS RAW fence (same wave wrote Ps, now re-reads cross-lane)
    asm volatile("s_wait_dscnt 0x0" ::: "memory");

    // O += P * V : P is A (16x64, two K-steps), V^T tile gives dword B-frags
    FragA pa[2];
#pragma unroll
    for (int ks = 0; ks < 2; ++ks)
#pragma unroll
      for (int i = 0; i < 8; ++i)
        pa[ks].u[i] =
            *(const unsigned int*)&Ps[w * 1024 + l15 * 64 + ka_off(i, sel) + ks * 32];
#pragma unroll
    for (int ht = 0; ht < 4; ++ht)
#pragma unroll
      for (int ks = 0; ks < 2; ++ks) {
        FragA vf;
#pragma unroll
        for (int i = 0; i < 8; ++i)
          vf.u[i] = *(const unsigned int*)
              &Vs[ib][(ht * 16 + l15) * 64 + kb_off(i, sel) + ks * 32];
        o[ht] = __builtin_amdgcn_wmma_f32_16x16x32_bf16(false, pa[ks].v, false, vf.v,
                                                        (short)0, o[ht], false, false);
      }

    if (kt + 1 < ktiles) {
      wait_async0();
      __syncthreads();
    }
  }

  // normalize and emit attn_out (bf16, [B*S][32*64])
#pragma unroll
  for (int ht = 0; ht < 4; ++ht)
#pragma unroll
    for (int j = 0; j < 8; ++j) {
      size_t row = rowQ + w * 16 + j + 8 * sel;
      int col = head * 64 + ht * 16 + l15;
      Out[row * 2048 + col] = f2bf(o[ht][j] * (1.0f / rsum[j]));
    }
}

// ---------------------------------------------------------------------------
// Workspace layout (bytes):
//  [0,        16777216)  residual bf16    4096x2048
//  [16777216, 29360128)  W_{Q,K,V} bf16   3072x2048 (Q rows 0-2047, K 2048-2559, V 2560-3071)
//  [29360128, 37748736)  W_O^T bf16       2048x2048
//  [37748736, 62914560)  QKV bf16         4096x3072
//  [62914560, 79691776)  attn_out bf16    4096x2048
//  [79691776, 83886080)  V^T bf16         2x8x64x2048
// ---------------------------------------------------------------------------
extern "C" void kernel_launch(void* const* d_in, const int* in_sizes, int n_in,
                              void* d_out, int out_size, void* d_ws, size_t ws_size,
                              hipStream_t stream) {
  (void)in_sizes; (void)n_in; (void)out_size; (void)ws_size;
  const float* residual = (const float*)d_in[0];
  const float* Wq = (const float*)d_in[1];
  const float* Wk = (const float*)d_in[2];
  const float* Wv = (const float*)d_in[3];
  const float* Wo = (const float*)d_in[4];

  char* ws = (char*)d_ws;
  unsigned short* Xb   = (unsigned short*)(ws);
  unsigned short* Wqkv = (unsigned short*)(ws + 16777216);
  unsigned short* Wot  = (unsigned short*)(ws + 29360128);
  unsigned short* QKVb = (unsigned short*)(ws + 37748736);
  unsigned short* Attb = (unsigned short*)(ws + 62914560);
  unsigned short* Vtb  = (unsigned short*)(ws + 79691776);

  cast_f32_to_bf16<<<1024, 256, 0, stream>>>(residual, Xb, 4096 * 2048);
  cast_f32_to_bf16<<<1024, 256, 0, stream>>>(Wq, Wqkv, 2048 * 2048);
  cast_f32_to_bf16<<<512,  256, 0, stream>>>(Wk, Wqkv + 2048 * 2048, 512 * 2048);
  cast_f32_to_bf16<<<512,  256, 0, stream>>>(Wv, Wqkv + 2560 * 2048, 512 * 2048);
  cast_transpose_wo<<<4096, 256, 0, stream>>>(Wo, Wot);

  // QKV projection: [4096x2048] x [3072x2048]^T -> [4096x3072]
  gemm_wmma_bf16<true><<<dim3(24, 64), 256, 0, stream>>>(Xb, Wqkv, QKVb, 2048, 3072);
  // V -> V^T (per batch, per kv head)
  transpose_v<<<4096, 256, 0, stream>>>(QKVb, Vtb);
  // causal GQA attention
  flash_attn_wmma<<<dim3(16, 32, 2), 256, 0, stream>>>(QKVb, Vtb, Attb);
  // output projection: [4096x2048] x [2048x2048]^T -> fp32 d_out
  gemm_wmma_bf16<false><<<dim3(16, 64), 256, 0, stream>>>(Attb, Wot, d_out, 2048, 2048);
}
